// MHSA_16784732192972
// MI455X (gfx1250) — matrix operand order repaired and checked by reading, not compiled, not run on hardware
//
#include <hip/hip_runtime.h>

#define DM 512
#define NH 8
#define DH 64
#define BATCH 2
#define SEQ 4096

typedef __attribute__((ext_vector_type(16))) _Float16 v16h;
typedef __attribute__((ext_vector_type(8)))  _Float16 v8h;
typedef __attribute__((ext_vector_type(8)))  float    v8f;

static __device__ __forceinline__ v16h load_a16(const _Float16* p0, const _Float16* p1) {
    v16h r;
    *(v8h*)&r       = *(const v8h*)p0;
    *((v8h*)&r + 1) = *(const v8h*)p1;
    return r;
}

static __device__ __forceinline__ v16h cvt_a16(const float* p0, const float* p1) {
    v16h r;
#pragma unroll
    for (int i = 0; i < 8; ++i) {
        r[i]     = (_Float16)p0[i];
        r[8 + i] = (_Float16)p1[i];
    }
    return r;
}

static __device__ __forceinline__ v16h gather16(const _Float16* p, int stride, int g0) {
    v16h r;
#pragma unroll
    for (int i = 0; i < 8; ++i) {
        r[i]     = p[(g0 + i) * stride];
        r[8 + i] = p[(g0 + 16 + i) * stride];
    }
    return r;
}

static __device__ __forceinline__ v8f wmma_f16(v16h a, v16h b, v8f c) {
    return __builtin_amdgcn_wmma_f32_16x16x32_f16(false, a, false, b, (short)0, c, false, false);
}

// Async DMA: global -> LDS, 16 bytes per lane. VDST carries the LDS byte
// offset (low 32 bits of the flat shared pointer), tracked by ASYNCcnt.
static __device__ __forceinline__ void async_b128(unsigned lds_off, const void* gaddr) {
    asm volatile("global_load_async_to_lds_b128 %0, %1, off"
                 :: "v"(lds_off), "v"(gaddr) : "memory");
}

// ---------------------------------------------------------------------------
// Kernel 1: transpose weights [n][k] f32 -> [k][n] f16 (WMMA B-fragment friendly)
// ---------------------------------------------------------------------------
__global__ __launch_bounds__(256) void cvt_w(const float* __restrict__ W0, const float* __restrict__ W1,
                                             const float* __restrict__ W2, const float* __restrict__ W3,
                                             _Float16* __restrict__ D0, _Float16* __restrict__ D1,
                                             _Float16* __restrict__ D2, _Float16* __restrict__ D3) {
    int idx = blockIdx.x * 256 + threadIdx.x;      // 0 .. 4*512*512-1
    int w   = idx >> 18;
    int r   = idx & 262143;
    int n   = r >> 9;
    int k   = r & 511;
    const float* S = (w == 0) ? W0 : (w == 1) ? W1 : (w == 2) ? W2 : W3;
    _Float16*    D = (w == 0) ? D0 : (w == 1) ? D1 : (w == 2) ? D2 : D3;
    D[k * DM + n] = (_Float16)S[n * DM + k];
}

// ---------------------------------------------------------------------------
// Kernel 2: projection  X(f32)[8192,512] @ WT(f16)[512,512] + bias -> f16 heads
// mode 0: dst[b,h,s,d]   mode 1: dst[b,h,d,s] (transposed K for QK^T B-frags)
// ---------------------------------------------------------------------------
__global__ __launch_bounds__(256) void proj_qkv(const float* __restrict__ X, const _Float16* __restrict__ WT,
                                                const float* __restrict__ bias, _Float16* __restrict__ dst,
                                                int mode) {
    const int lane = threadIdx.x & 31;
    const int wv   = threadIdx.x >> 5;
    const int task = blockIdx.x * 8 + wv;          // 0..4095
    const int mt   = task >> 3;                    // 0..511
    const int nt   = task & 7;                     // 0..7
    const int m0   = mt * 16;
    const int n0   = nt * 64;
    const int row  = m0 + (lane & 15);
    const int g0   = (lane < 16) ? 0 : 8;

    v8f acc[4] = {};
    for (int kb = 0; kb < DM; kb += 32) {
        v16h a = cvt_a16(X + (long)row * DM + kb + g0, X + (long)row * DM + kb + g0 + 16);
        const _Float16* wrow = WT + (long)kb * DM + n0 + (lane & 15);   // B col n = lane&15, k strided
#pragma unroll
        for (int t = 0; t < 4; ++t) {
            v16h b = gather16(wrow + t * 16, DM, g0);
            acc[t] = wmma_f16(a, b, acc[t]);
        }
    }

    const int nlane = lane & 15;
    const int rsel  = (lane < 16) ? 0 : 8;
#pragma unroll
    for (int t = 0; t < 4; ++t) {
        int ng = n0 + t * 16 + nlane;
        int h  = ng >> 6;
        int d  = ng & 63;
#pragma unroll
        for (int j = 0; j < 8; ++j) {
            int  m  = m0 + j + rsel;
            int  bb = m >> 12;                     // batch
            int  s  = m & 4095;                    // seq pos
            float val = acc[t][j] + bias[ng];
            long off = (mode == 1) ? ((long)((bb * NH + h) * DH + d)) * SEQ + s
                                   : ((long)(bb * NH + h) * SEQ + s) * DH + d;
            dst[off] = (_Float16)val;
        }
    }
}

// ---------------------------------------------------------------------------
// Kernel 3: flash attention. Block = 8 waves sharing one (b,h); each wave owns
// a 16-query tile. K^T/V tiles are staged into LDS with double-buffered
// global_load_async_to_lds_b128 (ASYNCcnt) so the DMA of chunk i+1 overlaps
// the 8 WMMAs + online softmax of chunk i.
// ---------------------------------------------------------------------------
__global__ __launch_bounds__(256) void attn(const _Float16* __restrict__ Qh, const _Float16* __restrict__ KhT,
                                            const _Float16* __restrict__ Vh, _Float16* __restrict__ ctx) {
    __shared__ _Float16 ktile[2][64 * 32];         // [d][key]  4KB per buffer
    __shared__ _Float16 vtile[2][32 * 64];         // [key][d]  4KB per buffer
    __shared__ _Float16 plds[8][16 * 32];          // per-wave P staging

    const int tid  = threadIdx.x;
    const int lane = tid & 31;
    const int wv   = tid >> 5;
    const int bh   = blockIdx.x >> 5;              // 0..15 : all 8 waves share head
    const int qb   = blockIdx.x & 31;              // 128-query block
    const int q0   = qb * 128 + wv * 16;
    const int qrow = lane & 15;
    const int g0   = (lane < 16) ? 0 : 8;

    const _Float16* ktbase = KhT + (long)bh * DH * SEQ;
    const _Float16* vbase  = Vh + (long)bh * SEQ * DH;

    // Per-thread async-copy slice: 256 threads x 16B cover each 4KB tile.
    const int kt_d = tid >> 2;                     // 0..63
    const int kt_k = (tid & 3) * 8;                // 0,8,16,24
    const int v_k  = tid >> 3;                     // 0..31
    const int v_d  = (tid & 7) * 8;                // 0..56
    unsigned kt_lds[2], v_lds[2];
    kt_lds[0] = (unsigned)(unsigned long long)&ktile[0][kt_d * 32 + kt_k];
    kt_lds[1] = (unsigned)(unsigned long long)&ktile[1][kt_d * 32 + kt_k];
    v_lds[0]  = (unsigned)(unsigned long long)&vtile[0][v_k * 64 + v_d];
    v_lds[1]  = (unsigned)(unsigned long long)&vtile[1][v_k * 64 + v_d];
    const _Float16* kt_g = ktbase + (long)kt_d * SEQ + kt_k;   // + kb
    const _Float16* v_g  = vbase + (long)v_k * DH + v_d;       // + kb*DH

    const long qbase = ((long)bh * SEQ + q0 + qrow) * DH;
    v16h qa0 = load_a16(Qh + qbase + g0,      Qh + qbase + g0 + 16);       // d 0..31
    v16h qa1 = load_a16(Qh + qbase + 32 + g0, Qh + qbase + 32 + g0 + 16);  // d 32..63

    v8f acc[4] = {};
    float rm[8], rl[8];
#pragma unroll
    for (int j = 0; j < 8; ++j) { rm[j] = -1e30f; rl[j] = 0.f; }

    _Float16* pl = &plds[wv][0];

    // prologue: DMA first K/V chunk into buffer 0
    async_b128(kt_lds[0], kt_g);
    async_b128(v_lds[0], v_g);

    int buf = 0;
    for (int kb = 0; kb < SEQ; kb += 32) {
        if (kb + 32 < SEQ) {                       // DMA next chunk, overlap compute
            async_b128(kt_lds[buf ^ 1], kt_g + (kb + 32));
            async_b128(v_lds[buf ^ 1], v_g + (long)(kb + 32) * DH);
            asm volatile("s_wait_asynccnt 0x2" ::: "memory");
        } else {
            asm volatile("s_wait_asynccnt 0x0" ::: "memory");
        }
        __syncthreads();                           // current buffer ready block-wide

        // --- scores: two 16x16 tiles (keys kb..+15, kb+16..+31) ---
        const _Float16* kr0 = &ktile[buf][lane & 15];             // B col key = lane&15, d 0..31
        const _Float16* kr1 = &ktile[buf][32 * 32 + (lane & 15)]; // B col key = lane&15, d 32..63
        v8f s0 = {}, s1 = {};
        s0 = wmma_f16(qa0, gather16(kr0, 32, g0), s0);
        s0 = wmma_f16(qa1, gather16(kr1, 32, g0), s0);
        s1 = wmma_f16(qa0, gather16(kr0 + 16, 32, g0), s1);
        s1 = wmma_f16(qa1, gather16(kr1 + 16, 32, g0), s1);

        // --- online softmax (row reductions stay inside 16-lane halves) ---
#pragma unroll
        for (int j = 0; j < 8; ++j) {
            float x0 = s0[j] * 0.125f;             // 1/sqrt(64)
            float x1 = s1[j] * 0.125f;
            float tmax = fmaxf(x0, x1);
            tmax = fmaxf(tmax, __shfl_xor(tmax, 1, 32));
            tmax = fmaxf(tmax, __shfl_xor(tmax, 2, 32));
            tmax = fmaxf(tmax, __shfl_xor(tmax, 4, 32));
            tmax = fmaxf(tmax, __shfl_xor(tmax, 8, 32));
            float mN   = fmaxf(rm[j], tmax);
            float corr = __expf(rm[j] - mN);
            float p0   = __expf(x0 - mN);
            float p1   = __expf(x1 - mN);
            float rs   = p0 + p1;
            rs += __shfl_xor(rs, 1, 32);
            rs += __shfl_xor(rs, 2, 32);
            rs += __shfl_xor(rs, 4, 32);
            rs += __shfl_xor(rs, 8, 32);
            rl[j] = rl[j] * corr + rs;
            rm[j] = mN;
#pragma unroll
            for (int t4 = 0; t4 < 4; ++t4) acc[t4][j] *= corr;
            int mrow = j + ((lane < 16) ? 0 : 8);
            pl[mrow * 32 + (lane & 15)]      = (_Float16)p0;
            pl[mrow * 32 + (lane & 15) + 16] = (_Float16)p1;
        }

        // --- reload P as A-fragment (same-wave DS ops are in order) ---
        v16h pa = load_a16(pl + qrow * 32 + g0, pl + qrow * 32 + g0 + 16);

        // --- acc += P @ V  (B col n = d lane&15, k = key strided) ---
        const _Float16* vr = &vtile[buf][lane & 15];
#pragma unroll
        for (int t = 0; t < 4; ++t)
            acc[t] = wmma_f16(pa, gather16(vr + t * 16, 64, g0), acc[t]);

        __syncthreads();                           // done consuming before overwrite
        buf ^= 1;
    }

    // --- normalize & write ctx[b, s, h*64+d] (model-dim contiguous) ---
    const int bb = bh >> 3, h = bh & 7;
#pragma unroll
    for (int j = 0; j < 8; ++j) {
        float inv = 1.f / rl[j];
        int   m   = q0 + j + ((lane < 16) ? 0 : 8);
        long  base = ((long)bb * SEQ + m) * DM + h * DH;
#pragma unroll
        for (int t = 0; t < 4; ++t)
            ctx[base + t * 16 + (lane & 15)] = (_Float16)(acc[t][j] * inv);
    }
}

// ---------------------------------------------------------------------------
// Kernel 4: output projection  ctx(f16)[8192,512] @ WoT + bo -> f32 out
// ---------------------------------------------------------------------------
__global__ __launch_bounds__(256) void out_proj(const _Float16* __restrict__ X, const _Float16* __restrict__ WT,
                                                const float* __restrict__ bias, float* __restrict__ out) {
    const int lane = threadIdx.x & 31;
    const int wv   = threadIdx.x >> 5;
    const int task = blockIdx.x * 8 + wv;
    const int mt   = task >> 3;
    const int nt   = task & 7;
    const int m0   = mt * 16;
    const int n0   = nt * 64;
    const int row  = m0 + (lane & 15);
    const int g0   = (lane < 16) ? 0 : 8;

    v8f acc[4] = {};
    for (int kb = 0; kb < DM; kb += 32) {
        v16h a = load_a16(X + (long)row * DM + kb + g0, X + (long)row * DM + kb + g0 + 16);
        const _Float16* wrow = WT + (long)kb * DM + n0 + (lane & 15);
#pragma unroll
        for (int t = 0; t < 4; ++t) {
            v16h b = gather16(wrow + t * 16, DM, g0);
            acc[t] = wmma_f16(a, b, acc[t]);
        }
    }

    const int nlane = lane & 15;
    const int rsel  = (lane < 16) ? 0 : 8;
#pragma unroll
    for (int t = 0; t < 4; ++t) {
        int ng = n0 + t * 16 + nlane;
#pragma unroll
        for (int j = 0; j < 8; ++j) {
            int m = m0 + j + rsel;
            out[(long)m * DM + ng] = acc[t][j] + bias[ng];
        }
    }
}

// ---------------------------------------------------------------------------
extern "C" void kernel_launch(void* const* d_in, const int* in_sizes, int n_in,
                              void* d_out, int out_size, void* d_ws, size_t ws_size,
                              hipStream_t stream) {
    (void)in_sizes; (void)n_in; (void)out_size; (void)ws_size;
    const float* q  = (const float*)d_in[0];
    const float* k  = (const float*)d_in[1];
    const float* v  = (const float*)d_in[2];
    const float* Wq = (const float*)d_in[3];
    const float* bq = (const float*)d_in[4];
    const float* Wk = (const float*)d_in[5];
    const float* bk = (const float*)d_in[6];
    const float* Wv = (const float*)d_in[7];
    const float* bv = (const float*)d_in[8];
    const float* Wo = (const float*)d_in[9];
    const float* bo = (const float*)d_in[10];
    float* out = (float*)d_out;

    char* ws = (char*)d_ws;
    const size_t KB = 1024, MB = 1024 * 1024;
    _Float16* WqT = (_Float16*)(ws + 0 * 512 * KB);
    _Float16* WkT = (_Float16*)(ws + 1 * 512 * KB);
    _Float16* WvT = (_Float16*)(ws + 2 * 512 * KB);
    _Float16* WoT = (_Float16*)(ws + 3 * 512 * KB);
    _Float16* Qh  = (_Float16*)(ws + 2 * MB);           // [b,h,s,d]  8MB
    _Float16* KhT = (_Float16*)(ws + 2 * MB + 8 * MB);  // [b,h,d,s]  8MB
    _Float16* Vh  = (_Float16*)(ws + 2 * MB + 16 * MB); // [b,h,s,d]  8MB
    _Float16* Ctx = (_Float16*)(ws + 2 * MB + 24 * MB); // [b,s,dm]   8MB

    cvt_w<<<4096, 256, 0, stream>>>(Wq, Wk, Wv, Wo, WqT, WkT, WvT, WoT);
    proj_qkv<<<512, 256, 0, stream>>>(q, WqT, bq, Qh, 0);
    proj_qkv<<<512, 256, 0, stream>>>(k, WkT, bk, KhT, 1);
    proj_qkv<<<512, 256, 0, stream>>>(v, WvT, bv, Vh, 0);
    attn<<<512, 256, 0, stream>>>(Qh, KhT, Vh, Ctx);
    out_proj<<<512, 256, 0, stream>>>(Ctx, WoT, bo, out);
}
